// ClusterMemoryCenter_57921928954234
// MI455X (gfx1250) — matrix operand order repaired and checked by reading, not compiled, not run on hardware
//
#include <hip/hip_runtime.h>
#include <hip/hip_bf16.h>

// ---------------------------------------------------------------------------
// Contrastive loss:  logits = inputs @ features^T / 0.05, split-half CE.
// B=256 batch, D=256, 2N=131072 feature rows. Split-bf16 (hi+lo) WMMA GEMM,
// fused online-softmax epilogue, two-pass cross-workgroup LSE reduction.
// Round 2: software-pipelined A-fragment and B-tile loads (double buffering)
// so global loads overlap the WMMA chains instead of s_wait_loadcnt 0x0
// serialization observed in round-1 assembly.
// ---------------------------------------------------------------------------

typedef __attribute__((ext_vector_type(16))) __bf16 v16bf;
typedef __attribute__((ext_vector_type(8)))  float  v8f;

union Frag16 { v16bf v; unsigned short u[16]; };

__device__ __forceinline__ unsigned short f32_bf16_rne(float f) {
    unsigned int x = __float_as_uint(f);
    x += 0x7FFFu + ((x >> 16) & 1u);      // round-to-nearest-even
    return (unsigned short)(x >> 16);
}
__device__ __forceinline__ float bf16_f32(unsigned short h) {
    return __uint_as_float(((unsigned int)h) << 16);
}

// ---------------------------------------------------------------------------
// Kernel 1: pack A = inputs (256x256 f32) into WMMA A-fragments, split hi/lo.
// Fragment f = i*8 + kstep (i = M-tile 0..15, kstep 0..7), 32 lanes x 16 bf16.
// A 16x32 layout: lane m = lane&15; kbase = kstep*32 + (lane>=16?16:0);
//   K = kbase+e (contiguous, matches B).
// ---------------------------------------------------------------------------
__global__ void __launch_bounds__(32)
prep_a_kernel(const float* __restrict__ inputs,
              unsigned short* __restrict__ ahi,
              unsigned short* __restrict__ alo) {
    const int f     = blockIdx.x;        // 0..127
    const int i     = f >> 3;
    const int kstep = f & 7;
    const int lane  = threadIdx.x;       // 0..31
    const int m     = i * 16 + (lane & 15);
    const int kbase = kstep * 32 + ((lane & 16) ? 16 : 0);
    const int base  = f * 512 + lane * 16;
#pragma unroll
    for (int e = 0; e < 16; ++e) {
        const int k = kbase + e;
        const float x = inputs[m * 256 + k];
        const unsigned short hb = f32_bf16_rne(x);
        ahi[base + e] = hb;
        alo[base + e] = f32_bf16_rne(x - bf16_f32(hb));
    }
}

// ---------------------------------------------------------------------------
// Kernel 2: per-workgroup GEMM tile (256 rows x 128 cols) + online softmax.
// Wave w owns cols [blk*128 + w*16, +16). B fragment built on the fly from
// f32 features (row-major N x K == B^T), split hi/lo.
// B 32x16 layout: lane n = lane&15; K = (lane>=16 ? 16..31 : 0..15) contiguous.
// ---------------------------------------------------------------------------
__global__ void __launch_bounds__(256)
logits_partial_kernel(const float* __restrict__ features,
                      const int*   __restrict__ targets,
                      const unsigned short* __restrict__ ahi,
                      const unsigned short* __restrict__ alo,
                      float* __restrict__ pmax,
                      float* __restrict__ psum,
                      float* __restrict__ tlog) {
    __shared__ float lmax[256][8];
    __shared__ float lsum[256][8];
    __shared__ int   tgt[256];

    const int tid  = threadIdx.x;
    const int wave = tid >> 5;
    const int lane = tid & 31;
    tgt[tid] = targets[tid];
    __syncthreads();

    const int colbase = blockIdx.x * 128 + wave * 16;
    const int n       = colbase + (lane & 15);          // feature row (N index)
    const int khalf   = (lane & 16) ? 16 : 0;
    const float* brow = features + (size_t)n * 256 + khalf;
    const unsigned short* afrag_hi = ahi + lane * 16;
    const unsigned short* afrag_lo = alo + lane * 16;

    v8f acc[16];
#pragma unroll
    for (int i = 0; i < 16; ++i)
        acc[i] = (v8f){0.f,0.f,0.f,0.f,0.f,0.f,0.f,0.f};

    // ---- prologue: raw B tile for kstep 0 ----
    float4 q0, q1, q2, q3;
    {
        const float4* bp = (const float4*)brow;
        q0 = bp[0]; q1 = bp[1]; q2 = bp[2]; q3 = bp[3];
    }

    for (int kstep = 0; kstep < 8; ++kstep) {
        // convert current raw B tile -> split-bf16 fragments
        const float bv[16] = {q0.x,q0.y,q0.z,q0.w, q1.x,q1.y,q1.z,q1.w,
                              q2.x,q2.y,q2.z,q2.w, q3.x,q3.y,q3.z,q3.w};
        Frag16 bh, bl;
#pragma unroll
        for (int e = 0; e < 16; ++e) {
            const unsigned short hb = f32_bf16_rne(bv[e]);
            bh.u[e] = hb;
            bl.u[e] = f32_bf16_rne(bv[e] - bf16_f32(hb));
        }

        // issue next kstep's raw B loads early (clamped on last iteration)
        const int knext = (kstep < 7) ? (kstep + 1) : kstep;
        {
            const float4* bp = (const float4*)(brow + knext * 32);
            q0 = bp[0]; q1 = bp[1]; q2 = bp[2]; q3 = bp[3];
        }

        // ---- inner M loop, A-fragment double buffered ----
        v16bf ah = *(const v16bf*)(afrag_hi + (size_t)(0 * 8 + kstep) * 512);
        v16bf al = *(const v16bf*)(afrag_lo + (size_t)(0 * 8 + kstep) * 512);
#pragma unroll
        for (int i = 0; i < 16; ++i) {
            const int inext = (i < 15) ? (i + 1) : i;
            const v16bf ahn = *(const v16bf*)(afrag_hi + (size_t)(inext * 8 + kstep) * 512);
            const v16bf aln = *(const v16bf*)(afrag_lo + (size_t)(inext * 8 + kstep) * 512);
            // (ahi+alo)*(bhi+blo) ~= ahi*bhi + ahi*blo + alo*bhi
            acc[i] = __builtin_amdgcn_wmma_f32_16x16x32_bf16(
                         false, al, false, bh.v, (short)0, acc[i], false, false);
            acc[i] = __builtin_amdgcn_wmma_f32_16x16x32_bf16(
                         false, ah, false, bl.v, (short)0, acc[i], false, false);
            acc[i] = __builtin_amdgcn_wmma_f32_16x16x32_bf16(
                         false, ah, false, bh.v, (short)0, acc[i], false, false);
            ah = ahn;
            al = aln;
        }
    }

    // ---- epilogue: per-row max / sum-exp over this wave's 16 columns ----
    const float invT = 20.0f;                           // 1 / TEMP
    const bool  hard = (blockIdx.x >= 512);             // cols >= 65536
    const int   col  = colbase + (lane & 15);
#pragma unroll
    for (int i = 0; i < 16; ++i) {
        const int rbase = i * 16 + ((lane & 16) ? 8 : 0);
#pragma unroll
        for (int r = 0; r < 8; ++r) {
            const int row = rbase + r;
            const float logit = acc[i][r] * invT;
            const int t = tgt[row];
            const int tcol = hard ? (65536 + t) : t;
            if (col == tcol) tlog[(hard ? 256 : 0) + row] = logit;

            float m = logit;
#pragma unroll
            for (int off = 1; off < 16; off <<= 1)
                m = fmaxf(m, __shfl_xor(m, off, 32));
            float e = __expf(logit - m);
#pragma unroll
            for (int off = 1; off < 16; off <<= 1)
                e += __shfl_xor(e, off, 32);
            if ((lane & 15) == 0) { lmax[row][wave] = m; lsum[row][wave] = e; }
        }
    }
    __syncthreads();

    // combine the 8 waves' (max,sum) for row == tid, write workgroup partial
    float M = lmax[tid][0];
#pragma unroll
    for (int w = 1; w < 8; ++w) M = fmaxf(M, lmax[tid][w]);
    float S = 0.f;
#pragma unroll
    for (int w = 0; w < 8; ++w) S += lsum[tid][w] * __expf(lmax[tid][w] - M);
    pmax[blockIdx.x * 256 + tid] = M;
    psum[blockIdx.x * 256 + tid] = S;
}

// ---------------------------------------------------------------------------
// Kernel 3: fold 512 workgroup partials per half into LSE per row, compute
// the two cross-entropies and the final scalar loss.
// ---------------------------------------------------------------------------
__global__ void __launch_bounds__(256)
loss_reduce_kernel(const float* __restrict__ pmax,
                   const float* __restrict__ psum,
                   const float* __restrict__ tlog,
                   float* __restrict__ out) {
    __shared__ float s0[256];
    __shared__ float s1[256];
    const int b = threadIdx.x;
    float ce[2];
    for (int h = 0; h < 2; ++h) {
        const int w0 = h * 512;
        float M = -3.4e38f;
        for (int w = 0; w < 512; ++w)
            M = fmaxf(M, pmax[(size_t)(w0 + w) * 256 + b]);
        float S = 0.f;
        for (int w = 0; w < 512; ++w)
            S += psum[(size_t)(w0 + w) * 256 + b] *
                 __expf(pmax[(size_t)(w0 + w) * 256 + b] - M);
        const float lse = M + __logf(S);
        ce[h] = lse - tlog[h * 256 + b];
    }
    s0[b] = ce[0];   // mean half
    s1[b] = ce[1];   // hard half
    __syncthreads();
    for (int stride = 128; stride > 0; stride >>= 1) {
        if (b < stride) { s0[b] += s0[b + stride]; s1[b] += s1[b + stride]; }
        __syncthreads();
    }
    if (b == 0) {
        const float ce_mean = s0[0] * (1.0f / 256.0f);
        const float ce_hard = s1[0] * (1.0f / 256.0f);
        out[0] = 0.5f * (ce_hard + fmaxf(ce_mean - 0.2f, 0.f));
    }
}

// ---------------------------------------------------------------------------
extern "C" void kernel_launch(void* const* d_in, const int* in_sizes, int n_in,
                              void* d_out, int out_size, void* d_ws, size_t ws_size,
                              hipStream_t stream) {
    const float* inputs   = (const float*)d_in[0];  // (256,256) f32
    const int*   targets  = (const int*)  d_in[1];  // (256,)
    const float* features = (const float*)d_in[2];  // (131072,256) f32

    char* ws = (char*)d_ws;
    unsigned short* ahi = (unsigned short*)(ws);                     // 128 KB
    unsigned short* alo = (unsigned short*)(ws + 131072);            // 128 KB
    float* pmax = (float*)(ws + 262144);                             // 1 MB
    float* psum = (float*)(ws + 262144 + 1048576);                   // 1 MB
    float* tlog = (float*)(ws + 262144 + 2097152);                   // 2 KB

    prep_a_kernel<<<128, 32, 0, stream>>>(inputs, ahi, alo);
    logits_partial_kernel<<<1024, 256, 0, stream>>>(features, targets,
                                                    ahi, alo, pmax, psum, tlog);
    loss_reduce_kernel<<<1, 256, 0, stream>>>(pmax, psum, tlog, (float*)d_out);
}
